// MambaLayer2D_1992864825682
// MI455X (gfx1250) — compile-verified, matrix operand order checked by reading, not run on hardware
//
#include <hip/hip_runtime.h>
#include <hip/hip_bf16.h>
#include <math.h>

#define DIM      256
#define D_STATE  16
#define D_CONV   4
#define D_INNER  512
#define DT_RANK  16
#define B_SZ     2
#define L_SEQ    4096
#define NROW     (B_SZ * L_SEQ)   /* 8192 rows of the "sequence x feature" matrices */
#define N_XZ     (2 * D_INNER)    /* 1024 */
#define N_DBL    (DT_RANK + 2 * D_STATE)  /* 48 */

typedef __bf16 bf16;
typedef __attribute__((ext_vector_type(16))) __bf16 bf16x16;
typedef __attribute__((ext_vector_type(8)))  __bf16 bf16x8;
typedef __attribute__((ext_vector_type(8)))  float  floatx8;

// ---- float -> bf16 (round-to-nearest-even), bit-level to avoid relying on fptrunc-to-bfloat ----
__device__ __forceinline__ bf16 f2bf(float f) {
    unsigned u = __float_as_uint(f);
    unsigned r = (u + 0x7FFFu + ((u >> 16) & 1u)) >> 16;
    unsigned short us = (unsigned short)r;
    bf16 out;
    __builtin_memcpy(&out, &us, 2);
    return out;
}

// ---- async global -> LDS 16-byte copy (CDNA5, tracked by ASYNCcnt) ----
// lds_byte_off is relative to the wave's LDS allocation; our kernels declare a
// single __shared__ object so its base offset is 0.
__device__ __forceinline__ void async_copy_b128(unsigned lds_byte_off, const bf16* gsrc) {
    asm volatile("global_load_async_to_lds_b128 %0, %1, off"
                 :: "v"(lds_byte_off), "v"(gsrc)
                 : "memory");
}
__device__ __forceinline__ void wait_async0() {
    asm volatile("s_wait_asynccnt 0x0" ::: "memory");
}

// ---- WMMA fragment loader from an LDS tile (row-major [rows][32] bf16) ----
// ISA 7.12.2 16-bit A/B layout: lane (idx = lane&15, h = lane>>4):
//   elements 0..7  <- K = 8h .. 8h+7 ;  elements 8..15 <- K = 16+8h .. 16+8h+7
__device__ __forceinline__ bf16x16 lds_frag(const bf16* __restrict__ tile, int row, int h) {
    const bf16* p = tile + row * 32 + 8 * h;
    bf16x8 lo = *(const bf16x8*)(p);
    bf16x8 hi = *(const bf16x8*)(p + 16);
    return __builtin_shufflevector(lo, hi, 0,1,2,3,4,5,6,7,8,9,10,11,12,13,14,15);
}

// ============================ weight transpose + bf16 convert ============================
__global__ void prep_weights_kernel(const float* __restrict__ Win, const float* __restrict__ Wx,
                                    const float* __restrict__ Wout,
                                    bf16* __restrict__ WinT, bf16* __restrict__ WxT,
                                    bf16* __restrict__ WoutT) {
    int idx = blockIdx.x * blockDim.x + threadIdx.x;
    const int n1 = N_XZ * DIM;       // WinT: [1024][256]
    const int n2 = N_DBL * D_INNER;  // WxT : [48][512]
    const int n3 = DIM * D_INNER;    // WoutT:[256][512]
    if (idx < n1) {
        int n = idx / DIM, k = idx % DIM;
        WinT[idx] = f2bf(Win[(size_t)k * N_XZ + n]);
    } else if (idx < n1 + n2) {
        int i = idx - n1;
        int n = i / D_INNER, k = i % D_INNER;
        WxT[i] = f2bf(Wx[(size_t)k * N_DBL + n]);
    } else if (idx < n1 + n2 + n3) {
        int i = idx - n1 - n2;
        int n = i / D_INNER, k = i % D_INNER;
        WoutT[i] = f2bf(Wout[(size_t)k * DIM + n]);
    }
}

// ============================ LayerNorm over channels, bf16 out ============================
__global__ void layernorm_kernel(const float* __restrict__ x, const float* __restrict__ g,
                                 const float* __restrict__ bta, bf16* __restrict__ xs) {
    __shared__ float s1[DIM];
    __shared__ float s2[DIM];
    int r = blockIdx.x;                 // r = b*L + l
    int c = threadIdx.x;
    int b = r >> 12, l = r & (L_SEQ - 1);
    float v = x[((size_t)b * DIM + c) * L_SEQ + l];
    s1[c] = v; s2[c] = v * v;
    __syncthreads();
    for (int s = DIM / 2; s > 0; s >>= 1) {
        if (c < s) { s1[c] += s1[c + s]; s2[c] += s2[c + s]; }
        __syncthreads();
    }
    float mu   = s1[0] * (1.f / DIM);
    float var  = s2[0] * (1.f / DIM) - mu * mu;
    float rstd = rsqrtf(var + 1e-5f);
    xs[(size_t)r * DIM + c] = f2bf((v - mu) * rstd * g[c] + bta[c]);
}

// ================= bf16 WMMA GEMM with async-LDS double buffering =================
// A: [M][K] bf16 row-major, Bt: [N][K] bf16 (weights transposed), C fp32.
// Block = 256 threads (8 waves) -> 128 rows x (NT*16) cols.  K must be a multiple of 64.
// Per K-step (32): stage A tile 128x32 + B tile (NT*16)x32 in LDS via
// global_load_async_to_lds_b128; consume via ds_load-fed v_wmma.  Ping/pong buffers
// with compile-time offsets; per-thread stage addresses hoisted out of the K loop.
template<int NT, bool SCATTER>
__global__ void wmma_gemm_async_kernel(const bf16* __restrict__ A, const bf16* __restrict__ Bt,
                                       float* __restrict__ C, int M, int N, int K) {
    constexpr int      A_ELEMS   = 128 * 32;
    constexpr int      B_ELEMS   = NT * 16 * 32;
    constexpr int      BUF_ELEMS = A_ELEMS + B_ELEMS;
    constexpr unsigned BUF_BYTES = (unsigned)BUF_ELEMS * 2u;
    __shared__ alignas(16) bf16 smem[2 * BUF_ELEMS];   // only shared object -> LDS offset 0

    const int tid  = threadIdx.x;
    const int lane = tid & 31;
    const int wave = tid >> 5;
    const int h    = lane >> 4;
    const int m0   = blockIdx.x * 128;
    const int n0   = blockIdx.y * (NT * 16);

    // ---- hoisted per-thread stage addresses (chunk = 16B; A: 512 chunks, B: NT*64) ----
    const int      rowA0 = tid >> 2;                 // chunks tid and tid+256
    const int      rowA1 = rowA0 + 64;
    const int      colA  = (tid & 3) * 8;            // elements
    const bf16*    gA0   = A + (size_t)(m0 + rowA0) * K + colA;
    const bf16*    gA1   = A + (size_t)(m0 + rowA1) * K + colA;
    const unsigned lA0   = (unsigned)(rowA0 * 32 + colA) * 2u;
    const unsigned lA1   = (unsigned)(rowA1 * 32 + colA) * 2u;
    const bool     hasB  = tid < NT * 64;
    const int      rowB  = tid >> 2;
    const bf16*    gB    = Bt + (size_t)(n0 + (hasB ? rowB : 0)) * K + colA;
    const unsigned lB    = (unsigned)(A_ELEMS * 2) + (unsigned)(rowB * 32 + colA) * 2u;

    auto issue = [&](unsigned bufb, int k0) {
        async_copy_b128(bufb + lA0, gA0 + k0);
        async_copy_b128(bufb + lA1, gA1 + k0);
        if (hasB) async_copy_b128(bufb + lB, gB + k0);
    };

    floatx8 acc[NT];
    #pragma unroll
    for (int t = 0; t < NT; ++t) acc[t] = floatx8{};

    auto compute = [&](const bf16* tileBase) {
        const bf16* tA = tileBase;
        const bf16* tB = tileBase + A_ELEMS;
        bf16x16 a = lds_frag(tA, wave * 16 + (lane & 15), h);
        bf16x16 bfr[NT];
        #pragma unroll
        for (int t = 0; t < NT; ++t)
            bfr[t] = lds_frag(tB, t * 16 + (lane & 15), h);
        #pragma unroll
        for (int t = 0; t < NT; ++t)
            acc[t] = __builtin_amdgcn_wmma_f32_16x16x32_bf16(
                false, a, false, bfr[t], (short)0, acc[t], false, false);
    };

    issue(0u, 0);
    for (int k0 = 0; k0 < K; k0 += 64) {           // K % 64 == 0 for all our GEMMs
        wait_async0();                             // stage of buffer 0 landed
        __syncthreads();                           // all waves staged; buffer 1 free
        issue(BUF_BYTES, k0 + 32);                 // overlap DMA with math
        compute(smem);                             // buffer 0
        wait_async0();
        __syncthreads();
        if (k0 + 64 < K) issue(0u, k0 + 64);
        compute(smem + BUF_ELEMS);                 // buffer 1
    }

    #pragma unroll
    for (int t = 0; t < NT; ++t) {
        #pragma unroll
        for (int j = 0; j < 8; ++j) {
            int row = m0 + wave * 16 + j + 8 * h;     // row = b*L + l
            int col = n0 + t * 16 + (lane & 15);
            if constexpr (SCATTER) {                  // store to (B, C, H, W)
                int b = row >> 12, l = row & (L_SEQ - 1);
                C[((size_t)b * DIM + col) * L_SEQ + l] = acc[t][j];
            } else {
                C[(size_t)row * N + col] = acc[t][j];
            }
        }
    }
}

// ============================ depthwise causal conv(4) + SiLU ============================
__global__ void conv_silu_kernel(const float* __restrict__ xz, const float* __restrict__ cw,
                                 const float* __restrict__ cb, float* __restrict__ xcf,
                                 bf16* __restrict__ xcb) {
    int idx = blockIdx.x * blockDim.x + threadIdx.x;
    if (idx >= NROW * D_INNER) return;
    int d = idx & (D_INNER - 1);
    int r = idx >> 9;
    int l = r & (L_SEQ - 1);
    float acc = cb[d];
    #pragma unroll
    for (int k = 0; k < D_CONV; ++k) {
        int lk = l + k - (D_CONV - 1);
        if (lk >= 0)
            acc += cw[d * D_CONV + k] * xz[(size_t)(r + k - (D_CONV - 1)) * N_XZ + d];
    }
    float s = acc / (1.f + __expf(-acc));   // SiLU
    xcf[idx] = s;
    xcb[idx] = f2bf(s);
}

// ============== dt = softplus(dbl[:, :16] @ W_dt + b_dt), K=16 on VALU ==============
__global__ void dt_kernel(const float* __restrict__ dbl, const float* __restrict__ Wdt,
                          const float* __restrict__ bdt, float* __restrict__ dt) {
    int idx = blockIdx.x * blockDim.x + threadIdx.x;
    if (idx >= NROW * D_INNER) return;
    int d = idx & (D_INNER - 1);
    int r = idx >> 9;
    float acc = bdt[d];
    #pragma unroll
    for (int j = 0; j < DT_RANK; ++j)
        acc += dbl[(size_t)r * N_DBL + j] * Wdt[j * D_INNER + d];
    dt[idx] = (acc > 20.f) ? acc : log1pf(__expf(acc));
}

// ============== selective scan + skip + SiLU gating, bf16 y ==============
// 4 blocks x 256 threads: one thread per (b, d); 16 states in registers; sequential in L.
__global__ void scan_kernel(const float* __restrict__ dt, const float* __restrict__ dbl,
                            const float* __restrict__ xcf, const float* __restrict__ xz,
                            const float* __restrict__ A_log, const float* __restrict__ Dskip,
                            bf16* __restrict__ y) {
    int b = blockIdx.x >> 1;
    int d = ((blockIdx.x & 1) << 8) + threadIdx.x;
    float A[D_STATE], h[D_STATE];
    #pragma unroll
    for (int n = 0; n < D_STATE; ++n) {
        A[n] = -__expf(A_log[d * D_STATE + n]);
        h[n] = 0.f;
    }
    float Dd = Dskip[d];
    for (int l = 0; l < L_SEQ; ++l) {
        int r = b * L_SEQ + l;
        if (l + 8 < L_SEQ) {   // stream-ahead hints -> global_prefetch_b8
            __builtin_prefetch(&dt [(size_t)(r + 8) * D_INNER + d], 0, 1);
            __builtin_prefetch(&xcf[(size_t)(r + 8) * D_INNER + d], 0, 1);
        }
        float dtv = dt[(size_t)r * D_INNER + d];
        float xv  = xcf[(size_t)r * D_INNER + d];
        const float* Bv = dbl + (size_t)r * N_DBL + DT_RANK;
        const float* Cv = Bv + D_STATE;
        float yv = 0.f;
        #pragma unroll
        for (int n = 0; n < D_STATE; ++n) {
            float dA = __expf(dtv * A[n]);
            h[n] = dA * h[n] + dtv * Bv[n] * xv;
            yv  += h[n] * Cv[n];
        }
        yv += xv * Dd;
        float zv = xz[(size_t)r * N_XZ + D_INNER + d];
        yv *= zv / (1.f + __expf(-zv));     // * silu(z)
        y[(size_t)r * D_INNER + d] = f2bf(yv);
    }
}

// =====================================================================================
extern "C" void kernel_launch(void* const* d_in, const int* in_sizes, int n_in,
                              void* d_out, int out_size, void* d_ws, size_t ws_size,
                              hipStream_t stream) {
    const float* x      = (const float*)d_in[0];
    const float* ln_g   = (const float*)d_in[1];
    const float* ln_b   = (const float*)d_in[2];
    const float* W_in   = (const float*)d_in[3];
    const float* conv_w = (const float*)d_in[4];
    const float* conv_b = (const float*)d_in[5];
    const float* W_x    = (const float*)d_in[6];
    const float* W_dt   = (const float*)d_in[7];
    const float* b_dt   = (const float*)d_in[8];
    const float* A_log  = (const float*)d_in[9];
    const float* D_skip = (const float*)d_in[10];
    const float* W_out  = (const float*)d_in[11];
    float* out = (float*)d_out;

    // ---- workspace carve-out (256-byte aligned) ----
    char* ws = (char*)d_ws;
    size_t off = 0;
    auto carve = [&](size_t bytes) -> void* {
        void* p = ws + off;
        off += (bytes + 255) & ~(size_t)255;
        return p;
    };
    bf16*  xs_bf   = (bf16*) carve((size_t)NROW * DIM * 2);        // 4 MB
    bf16*  WinT    = (bf16*) carve((size_t)N_XZ * DIM * 2);        // 0.5 MB
    bf16*  WxT     = (bf16*) carve((size_t)N_DBL * D_INNER * 2);   // 48 KB
    bf16*  WoutT   = (bf16*) carve((size_t)DIM * D_INNER * 2);     // 256 KB
    float* xz      = (float*)carve((size_t)NROW * N_XZ * 4);       // 32 MB
    float* xc_f    = (float*)carve((size_t)NROW * D_INNER * 4);    // 16 MB
    bf16*  xc_bf   = (bf16*) carve((size_t)NROW * D_INNER * 2);    // 8 MB
    float* dbl     = (float*)carve((size_t)NROW * N_DBL * 4);      // 1.5 MB
    float* dt      = (float*)carve((size_t)NROW * D_INNER * 4);    // 16 MB
    bf16*  y_bf    = (bf16*) carve((size_t)NROW * D_INNER * 2);    // 8 MB

    // 1) weights -> bf16 transposed
    {
        int total = N_XZ * DIM + N_DBL * D_INNER + DIM * D_INNER;
        prep_weights_kernel<<<(total + 255) / 256, 256, 0, stream>>>(
            W_in, W_x, W_out, WinT, WxT, WoutT);
    }
    // 2) layernorm -> xs_bf (8192 x 256)
    layernorm_kernel<<<NROW, DIM, 0, stream>>>(x, ln_g, ln_b, xs_bf);
    // 3) xz = xs @ W_in   (8192x256 * 256x1024)
    wmma_gemm_async_kernel<4, false><<<dim3(NROW / 128, N_XZ / 64), 256, 0, stream>>>(
        xs_bf, WinT, xz, NROW, N_XZ, DIM);
    // 4) depthwise causal conv + SiLU -> xc (fp32 + bf16)
    conv_silu_kernel<<<(NROW * D_INNER) / 256, 256, 0, stream>>>(
        xz, conv_w, conv_b, xc_f, xc_bf);
    // 5) dbl = xc @ W_x   (8192x512 * 512x48)
    wmma_gemm_async_kernel<3, false><<<dim3(NROW / 128, 1), 256, 0, stream>>>(
        xc_bf, WxT, dbl, NROW, N_DBL, D_INNER);
    // 6) dt = softplus(dbl[:, :16] @ W_dt + b_dt)
    dt_kernel<<<(NROW * D_INNER) / 256, 256, 0, stream>>>(dbl, W_dt, b_dt, dt);
    // 7) selective scan + skip + gating -> y_bf
    scan_kernel<<<B_SZ * (D_INNER / 256), 256, 0, stream>>>(
        dt, dbl, xc_f, xz, A_log, D_skip, y_bf);
    // 8) out = y @ W_out, scattered to (B, C, H, W)
    wmma_gemm_async_kernel<4, true><<<dim3(NROW / 128, DIM / 64), 256, 0, stream>>>(
        y_bf, WoutT, out, NROW, DIM, D_INNER);
}
